// MHA_39874476376765
// MI455X (gfx1250) — compile-verified, matrix operand order checked by reading, not run on hardware
//
#include <hip/hip_runtime.h>
#include <hip/hip_bf16.h>

// ---------------------------------------------------------------------------
// MHA forward for B=1, S=4096, D=1024, H=16, DH=64 on gfx1250 (CDNA5).
// Matrix math: v_wmma_f32_16x16x32_f16 (f16 in, f32 accumulate).
// Data movement: global_load_async_to_lds_b128 (ASYNCcnt) + double buffering.
// ---------------------------------------------------------------------------

#define S_LEN 4096
#define D_MODEL 1024
#define NHEAD 16
#define DHEAD 64
#define NEG_MASK (-10000.0f)

typedef __attribute__((ext_vector_type(16))) _Float16 v16h;
typedef __attribute__((ext_vector_type(8)))  float    v8f;

__device__ __forceinline__ v8f wmma_f16(v16h a, v16h b, v8f c) {
  // (neg_a, A, neg_b, B, c_mod, C, reuse_a, reuse_b)
  return __builtin_amdgcn_wmma_f32_16x16x32_f16(false, a, false, b,
                                                (short)0, c, false, false);
}

// Wave-relative LDS byte offset of a __shared__ object: the generic pointer's
// low 32 bits are the LDS address (ISA 10.2 aperture mapping truncates).
__device__ __forceinline__ uint32_t lds_off_of(const void* p) {
  return (uint32_t)(uintptr_t)p;
}

// One 16-byte async global->LDS transfer (GVS mode: SGPR base + 32-bit VGPR
// byte offset). Tracked by ASYNCcnt.
__device__ __forceinline__ void async_b128(const _Float16* smem_dst,
                                           const _Float16* gbase,
                                           uint32_t elem_off) {
  asm volatile("global_load_async_to_lds_b128 %0, %1, %2"
               :
               : "v"(lds_off_of(smem_dst)), "v"(elem_off * 2u), "s"(gbase)
               : "memory");
}

__device__ __forceinline__ void wait_async0() {
  asm volatile("s_wait_asynccnt 0x0" ::: "memory");
}

// A-fragment (16x32 f16) loader from a row-major [16][stride] tile.
// ISA 7.12.2: lanes 0-15 -> M=lane, K groups 0..7,16..23; lanes 16-31 -> +8.
// Also the B-fragment loader when the source tile is stored [N][K].
__device__ __forceinline__ v16h load_frag_rowK(const _Float16* p, int stride,
                                               int lane) {
  const int row = lane & 15;
  const int kg  = (lane >> 4) << 3;  // 0 or 8
  const _Float16* q = p + row * stride + kg;
  v16h f;
#pragma unroll
  for (int v = 0; v < 8; ++v) {
    const int kb = (v < 4) ? (2 * v) : (2 * v + 8);
    f[2 * v]     = q[kb];
    f[2 * v + 1] = q[kb + 1];
  }
  return f;
}

// B-fragment loader when the source tile is stored [K][ldn] (K-major).
__device__ __forceinline__ v16h load_frag_colK(const _Float16* p, int ldn,
                                               int lane) {
  const int col = lane & 15;
  const int kg  = (lane >> 4) << 3;
  v16h f;
#pragma unroll
  for (int v = 0; v < 8; ++v) {
    const int kb = ((v < 4) ? (2 * v) : (2 * v + 8)) + kg;
    f[2 * v]     = p[(size_t)kb * ldn + col];
    f[2 * v + 1] = p[(size_t)(kb + 1) * ldn + col];
  }
  return f;
}

// ---------------------------------------------------------------------------
// Kernel 0: one-shot f32 -> f16 conversion (x, Wqkv_w, out_w). ~20 MB once,
// ~1 us at 23.3 TB/s; makes every GEMM tile fill a raw async B128 copy.
// ---------------------------------------------------------------------------
__global__ __launch_bounds__(256) void cvt_f32_f16(const float* __restrict__ s,
                                                   _Float16* __restrict__ d,
                                                   int n) {
  const int i = (blockIdx.x * 256 + threadIdx.x) * 4;
  if (i + 3 < n) {
    d[i + 0] = (_Float16)s[i + 0];
    d[i + 1] = (_Float16)s[i + 1];
    d[i + 2] = (_Float16)s[i + 2];
    d[i + 3] = (_Float16)s[i + 3];
  }
}

// ---------------------------------------------------------------------------
// Kernel 1: QKV projection. out[s,e] = sum_d x[s,d] * Wqkv[e,d] + b[e]
// 128x64 tile per workgroup (256 threads = 8 waves, 4x2 of 32x32 per wave).
// Async-double-buffered LDS tiles. Writes Q/K/V f16 head-major [H][S][64];
// K pre-scaled by 1/sqrt(64).
// ---------------------------------------------------------------------------
__global__ __launch_bounds__(256) void mha_qkv_gemm(
    const _Float16* __restrict__ xh, const _Float16* __restrict__ wh,
    const float* __restrict__ bias, _Float16* __restrict__ qf,
    _Float16* __restrict__ kf, _Float16* __restrict__ vf) {
  __shared__ _Float16 Xs[2][128 * 32];
  __shared__ _Float16 Ws[2][64 * 32];
  const int tid  = threadIdx.x;
  const int lane = tid & 31;
  const int wv   = tid >> 5;          // 0..7
  const int wm   = (wv >> 1) * 32;    // 0,32,64,96
  const int wn   = (wv & 1) * 32;     // 0,32
  const int m0   = blockIdx.x * 128;
  const int n0   = blockIdx.y * 64;

  auto issue = [&](int buf, int k0) {
    {  // X tile: 128x32 halves, 16 per thread = 2 x b128
      const int i = tid * 16;
      const int r = i >> 5, c = i & 31;
      const uint32_t go = (uint32_t)((m0 + r) * D_MODEL + k0 + c);
      async_b128(&Xs[buf][i], xh, go);
      async_b128(&Xs[buf][i + 8], xh, go + 8);
    }
    {  // W tile: 64x32 halves, 8 per thread = 1 x b128
      const int i = tid * 8;
      const int r = i >> 5, c = i & 31;
      const uint32_t go = (uint32_t)((n0 + r) * D_MODEL + k0 + c);
      async_b128(&Ws[buf][i], wh, go);
    }
  };

  const v8f vzero = {0.f, 0.f, 0.f, 0.f, 0.f, 0.f, 0.f, 0.f};
  v8f acc[2][2];
  acc[0][0] = vzero; acc[0][1] = vzero; acc[1][0] = vzero; acc[1][1] = vzero;

  issue(0, 0);
  for (int kt = 0; kt < D_MODEL / 32; ++kt) {
    wait_async0();       // my async writes have landed
    __syncthreads();     // everyone's have; previous buffer free for reuse
    if (kt + 1 < D_MODEL / 32) issue((kt + 1) & 1, (kt + 1) * 32);

    const _Float16* Xc = Xs[kt & 1];
    const _Float16* Wc = Ws[kt & 1];
    const v16h a0 = load_frag_rowK(&Xc[(size_t)wm * 32], 32, lane);
    const v16h a1 = load_frag_rowK(&Xc[(size_t)(wm + 16) * 32], 32, lane);
    const v16h b0 = load_frag_rowK(&Wc[(size_t)wn * 32], 32, lane);
    const v16h b1 = load_frag_rowK(&Wc[(size_t)(wn + 16) * 32], 32, lane);
    acc[0][0] = wmma_f16(a0, b0, acc[0][0]);
    acc[0][1] = wmma_f16(a0, b1, acc[0][1]);
    acc[1][0] = wmma_f16(a1, b0, acc[1][0]);
    acc[1][1] = wmma_f16(a1, b1, acc[1][1]);
  }

  const int nlo = lane & 15;
  const int g   = lane >> 4;
#pragma unroll
  for (int i = 0; i < 2; ++i) {
#pragma unroll
    for (int j = 0; j < 2; ++j) {
#pragma unroll
      for (int r = 0; r < 8; ++r) {
        const int srow = m0 + wm + i * 16 + r + 8 * g;
        const int e    = n0 + wn + j * 16 + nlo;
        float val = acc[i][j][r] + bias[e];
        const int part = e >> 10;          // 0=q, 1=k, 2=v
        const int rem  = e & 1023;
        const int head = rem >> 6;
        const int dh   = rem & 63;
        if (part == 1) val *= 0.125f;      // softmax scale 1/sqrt(64)
        _Float16* dst = (part == 0) ? qf : (part == 1) ? kf : vf;
        dst[(size_t)head * S_LEN * DHEAD + (size_t)srow * DHEAD + dh] =
            (_Float16)val;
      }
    }
  }
}

// ---------------------------------------------------------------------------
// Kernel 2: causal flash attention. One workgroup (4 waves) per
// (head, 64 query rows); each wave owns 16 rows. Key blocks iterate only up
// to the causal diagonal; K/V tiles async-double-buffered in LDS.
// ---------------------------------------------------------------------------
__global__ __launch_bounds__(128) void mha_flash_attn(
    const _Float16* __restrict__ qf, const _Float16* __restrict__ kf,
    const _Float16* __restrict__ vf, _Float16* __restrict__ ctx) {
  __shared__ _Float16 Kt[2][64 * 64];
  __shared__ _Float16 Vt[2][64 * 64];
  __shared__ _Float16 Pst[4][16 * 64];

  const int tid  = threadIdx.x;
  const int lane = tid & 31;
  const int wv   = tid >> 5;               // 0..3
  const int h    = blockIdx.y;
  const int qb0  = blockIdx.x * 64;
  const int qrow = qb0 + wv * 16;          // this wave's first query row
  const size_t hbase = (size_t)h * S_LEN * DHEAD;
  const int nlo = lane & 15;
  const int g   = lane >> 4;

  const _Float16* kh = kf + hbase;
  const _Float16* vh = vf + hbase;

  auto issueKV = [&](int buf, int jb) {
    // 64x64 halves per tile = contiguous 4096 halves; 32 per thread = 4xb128
    const uint32_t base = (uint32_t)(jb * DHEAD) + (uint32_t)(tid * 32);
    const int i = tid * 32;
#pragma unroll
    for (int q4 = 0; q4 < 4; ++q4) {
      async_b128(&Kt[buf][i + q4 * 8], kh, base + q4 * 8);
      async_b128(&Vt[buf][i + q4 * 8], vh, base + q4 * 8);
    }
  };

  // Q fragments (K = dh = 64 -> two 16x32 fragments), kept in registers.
  const v16h qa0 =
      load_frag_rowK(qf + hbase + (size_t)qrow * DHEAD, DHEAD, lane);
  const v16h qa1 =
      load_frag_rowK(qf + hbase + (size_t)qrow * DHEAD + 32, DHEAD, lane);

  const v8f vzero = {0.f, 0.f, 0.f, 0.f, 0.f, 0.f, 0.f, 0.f};
  v8f o[4];
  float mrow[8], lrow[8];
#pragma unroll
  for (int t = 0; t < 4; ++t) o[t] = vzero;
#pragma unroll
  for (int r = 0; r < 8; ++r) { mrow[r] = -3.0e38f; lrow[r] = 0.f; }

  const int njb = (qb0 / 64) + 1;          // causal: key blocks <= diagonal
  issueKV(0, 0);
  for (int j = 0; j < njb; ++j) {
    const int jb = j * 64;
    wait_async0();
    __syncthreads();
    if (j + 1 < njb) issueKV((j + 1) & 1, (j + 1) * 64);

    const _Float16* Kc = Kt[j & 1];
    const _Float16* Vc = Vt[j & 1];

    // S = Q K^T  (16 rows x 64 keys as 4 column tiles)
    v8f sfr[4];
#pragma unroll
    for (int t = 0; t < 4; ++t) {
      const v16h kb0 = load_frag_rowK(&Kc[(size_t)(t * 16) * 64], 64, lane);
      const v16h kb1 =
          load_frag_rowK(&Kc[(size_t)(t * 16) * 64 + 32], 64, lane);
      sfr[t] = wmma_f16(qa0, kb0, vzero);
      sfr[t] = wmma_f16(qa1, kb1, sfr[t]);
    }

    // Causal mask (only the diagonal block can mask this wave's rows).
    if (jb + 63 > qrow) {
#pragma unroll
      for (int t = 0; t < 4; ++t) {
#pragma unroll
        for (int r = 0; r < 8; ++r) {
          const int kidx = jb + t * 16 + nlo;
          const int qidx = qrow + r + 8 * g;
          if (kidx > qidx) sfr[t][r] += NEG_MASK;
        }
      }
    }

    // Online softmax (rows live across 16-lane halves in C layout).
#pragma unroll
    for (int r = 0; r < 8; ++r) {
      float bm = fmaxf(fmaxf(sfr[0][r], sfr[1][r]),
                       fmaxf(sfr[2][r], sfr[3][r]));
#pragma unroll
      for (int off = 1; off < 16; off <<= 1)
        bm = fmaxf(bm, __shfl_xor(bm, off));
      const float mnew  = fmaxf(mrow[r], bm);
      const float alpha = __expf(mrow[r] - mnew);
      mrow[r] = mnew;
      lrow[r] *= alpha;
#pragma unroll
      for (int t = 0; t < 4; ++t) o[t][r] *= alpha;
      float rs = 0.f;
#pragma unroll
      for (int t = 0; t < 4; ++t) {
        const float p = __expf(sfr[t][r] - mnew);
        sfr[t][r] = p;
        rs += p;
      }
#pragma unroll
      for (int off = 1; off < 16; off <<= 1) rs += __shfl_xor(rs, off);
      lrow[r] += rs;
    }

    // C-layout -> A-layout for P via per-wave LDS staging (wave-local; LDS
    // ops from one wave stay in order, no barrier needed).
    _Float16* pw = &Pst[wv][0];
#pragma unroll
    for (int t = 0; t < 4; ++t)
#pragma unroll
      for (int r = 0; r < 8; ++r)
        pw[(size_t)(r + 8 * g) * 64 + t * 16 + nlo] = (_Float16)sfr[t][r];

    const v16h pa0 = load_frag_rowK(pw, 64, lane);
    const v16h pa1 = load_frag_rowK(pw + 32, 64, lane);
#pragma unroll
    for (int t = 0; t < 4; ++t) {
      const v16h vb0 = load_frag_colK(&Vc[t * 16], 64, lane);
      const v16h vb1 = load_frag_colK(&Vc[(size_t)32 * 64 + t * 16], 64, lane);
      o[t] = wmma_f16(pa0, vb0, o[t]);
      o[t] = wmma_f16(pa1, vb1, o[t]);
    }
  }

  // Normalize and write context f16 [S][D], column = h*64 + dh.
#pragma unroll
  for (int t = 0; t < 4; ++t) {
#pragma unroll
    for (int r = 0; r < 8; ++r) {
      const float val = o[t][r] / lrow[r];
      ctx[(size_t)(qrow + r + 8 * g) * D_MODEL + h * DHEAD + t * 16 + nlo] =
          (_Float16)val;
    }
  }
}

// ---------------------------------------------------------------------------
// Kernel 3: output projection. out[s,e] = sum_d ctx[s,d]*out_w[e,d] + b[e]
// Same async-double-buffered tiling as kernel 1; output f32 to d_out.
// ---------------------------------------------------------------------------
__global__ __launch_bounds__(256) void mha_out_gemm(
    const _Float16* __restrict__ ctx, const _Float16* __restrict__ wh,
    const float* __restrict__ bias, float* __restrict__ out) {
  __shared__ _Float16 Xs[2][128 * 32];
  __shared__ _Float16 Ws[2][64 * 32];
  const int tid  = threadIdx.x;
  const int lane = tid & 31;
  const int wv   = tid >> 5;
  const int wm   = (wv >> 1) * 32;
  const int wn   = (wv & 1) * 32;
  const int m0   = blockIdx.x * 128;
  const int n0   = blockIdx.y * 64;

  auto issue = [&](int buf, int k0) {
    {
      const int i = tid * 16;
      const int r = i >> 5, c = i & 31;
      const uint32_t go = (uint32_t)((m0 + r) * D_MODEL + k0 + c);
      async_b128(&Xs[buf][i], ctx, go);
      async_b128(&Xs[buf][i + 8], ctx, go + 8);
    }
    {
      const int i = tid * 8;
      const int r = i >> 5, c = i & 31;
      const uint32_t go = (uint32_t)((n0 + r) * D_MODEL + k0 + c);
      async_b128(&Ws[buf][i], wh, go);
    }
  };

  const v8f vzero = {0.f, 0.f, 0.f, 0.f, 0.f, 0.f, 0.f, 0.f};
  v8f acc[2][2];
  acc[0][0] = vzero; acc[0][1] = vzero; acc[1][0] = vzero; acc[1][1] = vzero;

  issue(0, 0);
  for (int kt = 0; kt < D_MODEL / 32; ++kt) {
    wait_async0();
    __syncthreads();
    if (kt + 1 < D_MODEL / 32) issue((kt + 1) & 1, (kt + 1) * 32);

    const _Float16* Xc = Xs[kt & 1];
    const _Float16* Wc = Ws[kt & 1];
    const v16h a0 = load_frag_rowK(&Xc[(size_t)wm * 32], 32, lane);
    const v16h a1 = load_frag_rowK(&Xc[(size_t)(wm + 16) * 32], 32, lane);
    const v16h b0 = load_frag_rowK(&Wc[(size_t)wn * 32], 32, lane);
    const v16h b1 = load_frag_rowK(&Wc[(size_t)(wn + 16) * 32], 32, lane);
    acc[0][0] = wmma_f16(a0, b0, acc[0][0]);
    acc[0][1] = wmma_f16(a0, b1, acc[0][1]);
    acc[1][0] = wmma_f16(a1, b0, acc[1][0]);
    acc[1][1] = wmma_f16(a1, b1, acc[1][1]);
  }

  const int nlo = lane & 15;
  const int g   = lane >> 4;
#pragma unroll
  for (int i = 0; i < 2; ++i)
#pragma unroll
    for (int j = 0; j < 2; ++j)
#pragma unroll
      for (int r = 0; r < 8; ++r) {
        const int srow = m0 + wm + i * 16 + r + 8 * g;
        const int e    = n0 + wn + j * 16 + nlo;
        out[(size_t)srow * D_MODEL + e] = acc[i][j][r] + bias[e];
      }
}

// ---------------------------------------------------------------------------
extern "C" void kernel_launch(void* const* d_in, const int* in_sizes, int n_in,
                              void* d_out, int out_size, void* d_ws,
                              size_t ws_size, hipStream_t stream) {
  (void)in_sizes; (void)n_in; (void)out_size; (void)ws_size;
  const float* x      = (const float*)d_in[0];
  const float* Wqkv_w = (const float*)d_in[1];
  const float* Wqkv_b = (const float*)d_in[2];
  const float* out_w  = (const float*)d_in[3];
  const float* out_b  = (const float*)d_in[4];
  float* out = (float*)d_out;

  // Workspace (f16): Q,K(scaled),V head-major [H][S][64] + ctx [S][D]
  // (4 x 4M halves = 32 MB), then xh (4M), wqkvh (3M), outwh (1M) = 48 MB.
  const size_t SD = (size_t)S_LEN * D_MODEL;
  _Float16* ws    = (_Float16*)d_ws;
  _Float16* qf    = ws;
  _Float16* kfp   = ws + SD;
  _Float16* vfp   = ws + 2 * SD;
  _Float16* ctx   = ws + 3 * SD;
  _Float16* xh    = ws + 4 * SD;
  _Float16* wqkvh = ws + 5 * SD;
  _Float16* outwh = ws + 8 * SD;

  const int nX = S_LEN * D_MODEL;          // 4194304
  const int nW = 3 * D_MODEL * D_MODEL;    // 3145728
  const int nO = D_MODEL * D_MODEL;        // 1048576
  cvt_f32_f16<<<nX / 1024, 256, 0, stream>>>(x, xh, nX);
  cvt_f32_f16<<<nW / 1024, 256, 0, stream>>>(Wqkv_w, wqkvh, nW);
  cvt_f32_f16<<<nO / 1024, 256, 0, stream>>>(out_w, outwh, nO);

  mha_qkv_gemm<<<dim3(S_LEN / 128, (3 * D_MODEL) / 64), 256, 0, stream>>>(
      xh, wqkvh, Wqkv_b, qf, kfp, vfp);
  mha_flash_attn<<<dim3(S_LEN / 64, NHEAD), 128, 0, stream>>>(qf, kfp, vfp,
                                                              ctx);
  mha_out_gemm<<<dim3(S_LEN / 128, D_MODEL / 64), 256, 0, stream>>>(
      ctx, outwh, out_b, out);
}
